// FlaxTPULlamaAttention_2688649528054
// MI455X (gfx1250) — matrix operand order repaired and checked by reading, not compiled, not run on hardware
//
#include <hip/hip_runtime.h>
#include <hip/hip_bf16.h>
#include <stdint.h>

#define SEQ     2048
#define HIDDEN  4096
#define NH      32
#define NKV     8
#define HD      128
#define NQKV    6144   /* 4096 Q + 1024 K + 1024 V columns */

typedef __attribute__((ext_vector_type(16))) __bf16 v16bf;
typedef __attribute__((ext_vector_type(8)))  __bf16 v8bf;
typedef __attribute__((ext_vector_type(8)))  float  v8f;

__device__ __forceinline__ unsigned short f2bf(float f) {
  unsigned u = __builtin_bit_cast(unsigned, f);
  u += 0x7FFFu + ((u >> 16) & 1u);            // round-to-nearest-even
  return (unsigned short)(u >> 16);
}
__device__ __forceinline__ float bf2f(unsigned short b) {
  unsigned u = ((unsigned)b) << 16;
  return __builtin_bit_cast(float, u);
}

// A-fragment (16x32 bf16, M x K): lane 0-15 -> M=lane, K chunks {0..7,16..23};
// lane 16-31 -> M=lane-16, K chunks {8..15,24..31}.  (ISA 7.12.2)
__device__ __forceinline__ v16bf load_a(const unsigned short* base, int stride, int lane) {
  int row = lane & 15;
  int kb  = (lane >> 4) << 3;                 // 0 or 8
  const v8bf* p0 = (const v8bf*)(base + row * stride + kb);
  const v8bf* p1 = (const v8bf*)(base + row * stride + kb + 16);
  v8bf lo = *p0, hi = *p1;
  return __builtin_shufflevector(lo, hi, 0,1,2,3,4,5,6,7,8,9,10,11,12,13,14,15);
}

// B-fragment (32x16 bf16, K x N) from an N-major (transposed) buffer:
// lane 0-15 -> N=lane, K chunks {0..7,16..23} ; lane 16-31 -> N=lane-16, K chunks {8..15,24..31}.
__device__ __forceinline__ v16bf load_b(const unsigned short* base, int stride, int lane) {
  int col = lane & 15;
  int kb  = (lane >> 4) << 3;                 // 0 or 8
  const v8bf* p0 = (const v8bf*)(base + col * stride + kb);
  const v8bf* p1 = p0 + 2;
  v8bf lo = *p0, hi = *p1;
  return __builtin_shufflevector(lo, hi, 0,1,2,3,4,5,6,7,8,9,10,11,12,13,14,15);
}

__device__ __forceinline__ v8f wmma_bf16(v16bf a, v16bf b, v8f c) {
  return __builtin_amdgcn_wmma_f32_16x16x32_bf16(false, a, false, b, (short)0, c,
                                                 false, false);
}

// CDNA5 async DMA: one b128 global -> LDS, tracked by ASYNCcnt (no VGPR data).
__device__ __forceinline__ void async_cp_b128(unsigned lds_off, unsigned long long gaddr) {
  asm volatile("global_load_async_to_lds_b128 %0, %1, off"
               :: "v"(lds_off), "v"(gaddr) : "memory");
}

// ---------------- conversion kernels ----------------
__global__ void cvt_x_kernel(const float* __restrict__ x, unsigned short* __restrict__ o, int n) {
  int i = blockIdx.x * blockDim.x + threadIdx.x;
  if (i < n) o[i] = f2bf(x[i]);
}

// W[K][N] (fp32) -> Wt[N][K] (bf16) so GEMM B fragments read contiguous K
__global__ void cvt_wT_kernel(const float* __restrict__ w, unsigned short* __restrict__ wt,
                              int K, int N) {
  int i = blockIdx.x * blockDim.x + threadIdx.x;
  if (i >= K * N) return;
  int k = i / N, n = i - k * N;
  wt[(size_t)n * K + k] = f2bf(w[i]);
}

// ---------------- shared GEMM tile engine ----------------
// Block (256 thr = 8 waves) computes a 128x64 tile: wave w owns rows
// [m0+16w, m0+16w+16).  The 64x32 B tile for each k-step is async-DMA'd into
// double-buffered LDS by the whole block (1 b128 per thread per step) and the
// A fragment is register double-buffered, so WMMAs overlap the async copies.
template <int KD>
__device__ __forceinline__ void gemm_wave_16x64(const unsigned short* __restrict__ Arow,
                                                const unsigned short* __restrict__ Bt,
                                                unsigned short* lds /* 2*64*32 bf16 */,
                                                int lane, int tid, v8f acc[4]) {
  const int NS = KD / 32;
  int crow = tid >> 2, cpart = tid & 3;                  // copy role: 4 b128 per B row
  const unsigned short* gsrc = Bt + (size_t)crow * KD + cpart * 8;
  unsigned myslot = (unsigned)(uintptr_t)lds + (unsigned)tid * 16u;  // low 32b = LDS offset

  async_cp_b128(myslot, (unsigned long long)(uintptr_t)gsrc);        // prologue: buf0,k=0
  v16bf a_cur = load_a(Arow, KD, lane);

  for (int i = 0; i < NS; ++i) {
    int k = i << 5;
    if (i + 1 < NS) {
      async_cp_b128(myslot + (unsigned)(((i + 1) & 1) << 12),
                    (unsigned long long)(uintptr_t)(gsrc + k + 32));
      asm volatile("s_wait_asynccnt 0x1" ::: "memory");  // step i landed (in-order)
    } else {
      asm volatile("s_wait_asynccnt 0x0" ::: "memory");
    }
    __syncthreads();                                     // publish B tile across waves

    v16bf a_nxt = a_cur;
    if (i + 1 < NS) a_nxt = load_a(Arow + k + 32, KD, lane);  // prefetch next A

    const unsigned short* bb = lds + (size_t)(i & 1) * (64 * 32);
    v16bf b0 = load_b(bb +   0, 32, lane);               // ds_load_b128 x2 each
    v16bf b1 = load_b(bb + 512, 32, lane);
    v16bf b2 = load_b(bb + 1024, 32, lane);
    v16bf b3 = load_b(bb + 1536, 32, lane);
    acc[0] = wmma_bf16(a_cur, b0, acc[0]);
    acc[1] = wmma_bf16(a_cur, b1, acc[1]);
    acc[2] = wmma_bf16(a_cur, b2, acc[2]);
    acc[3] = wmma_bf16(a_cur, b3, acc[3]);

    __syncthreads();                                     // tile consumed; buffer reusable
    a_cur = a_nxt;
  }
}

// ---------------- fused QKV projection ----------------
// X(2048x4096) @ [Wq|Wk|Wv] -> Q[h][s][d], K[h][s][d], V^T[h][d][s]  (bf16)
__global__ void __launch_bounds__(256) gemm_qkv_kernel(
    const unsigned short* __restrict__ X, const unsigned short* __restrict__ Wt,
    unsigned short* __restrict__ Q, unsigned short* __restrict__ K,
    unsigned short* __restrict__ Vt) {
  __shared__ __align__(16) unsigned short Bst[2][64][32];
  int tid = threadIdx.x, lane = tid & 31, wib = tid >> 5;
  int bm = blockIdx.x / (NQKV / 64);                     // 16 m-blocks
  int bn = blockIdx.x % (NQKV / 64);                     // 96 n-groups
  int m0 = bm * 128 + wib * 16;
  int n0 = bn * 64;

  v8f acc[4];
  v8f z = {};
  #pragma unroll
  for (int t = 0; t < 4; ++t) acc[t] = z;

  gemm_wave_16x64<HIDDEN>(X + (size_t)m0 * HIDDEN, Wt + (size_t)n0 * HIDDEN,
                          &Bst[0][0][0], lane, tid, acc);

  int l15 = lane & 15;
  int mb  = (lane >> 4) << 3;                 // C layout: M = elem + 8*(lane>=16)
  #pragma unroll
  for (int t = 0; t < 4; ++t) {
    #pragma unroll
    for (int j = 0; j < 8; ++j) {
      int m = m0 + mb + j;
      int n = n0 + t * 16 + l15;
      unsigned short bv = f2bf(acc[t][j]);
      if (n < 4096) {                         // Q: [h][s][d]
        int h = n >> 7, d = n & 127;
        Q[((size_t)h * SEQ + m) * HD + d] = bv;
      } else if (n < 5120) {                  // K: [h][s][d]
        int nn = n - 4096; int h = nn >> 7, d = nn & 127;
        K[((size_t)h * SEQ + m) * HD + d] = bv;
      } else {                                // V transposed: [h][d][s]
        int nn = n - 5120; int h = nn >> 7, d = nn & 127;
        Vt[((size_t)h * HD + d) * SEQ + m] = bv;
      }
    }
  }
}

// ---------------- RoPE (in place on Q and K) ----------------
__global__ void rope_kernel(unsigned short* __restrict__ Q, unsigned short* __restrict__ K,
                            const int* __restrict__ pos) {
  int row = blockIdx.x;                       // NH*SEQ Q rows then NKV*SEQ K rows
  int d   = threadIdx.x;                      // 0..63
  unsigned short* base;
  int s;
  if (row < NH * SEQ) { base = Q + (size_t)row * HD; s = row & (SEQ - 1); }
  else { int r = row - NH * SEQ; base = K + (size_t)r * HD; s = r & (SEQ - 1); }
  float p   = (float)pos[s];
  float inv = __powf(10000.0f, -(float)(2 * d) / (float)HD);
  float ang = p * inv;
  float c = __cosf(ang), sn = __sinf(ang);
  float x0 = bf2f(base[d]);
  float x1 = bf2f(base[d + 64]);
  base[d]      = f2bf(x0 * c - x1 * sn);
  base[d + 64] = f2bf(x1 * c + x0 * sn);
}

// ---------------- flash attention ----------------
// one wave per (head, 16-row q tile); kv tiles of 32; P re-fragmented via LDS
__global__ void __launch_bounds__(128) flash_attn_kernel(
    const unsigned short* __restrict__ Q, const unsigned short* __restrict__ K,
    const unsigned short* __restrict__ Vt, const int* __restrict__ amask,
    unsigned short* __restrict__ O) {
  __shared__ __align__(16) unsigned short Plds[4][16][32];
  int lane = threadIdx.x & 31;
  int wib  = threadIdx.x >> 5;
  int wave = blockIdx.x * 4 + wib;
  int h  = wave >> 7;                         // 128 q-tiles per head
  int qt = wave & 127;
  int q0 = qt << 4;
  int hk = h >> 2;                            // GQA: 4 Q heads per KV head
  int l15 = lane & 15;
  int mb  = (lane >> 4) << 3;

  v16bf qf[4];
  #pragma unroll
  for (int c = 0; c < 4; ++c)
    qf[c] = load_a(Q + ((size_t)h * SEQ + q0) * HD + c * 32, HD, lane);

  v8f acc[8];
  v8f z = {};
  #pragma unroll
  for (int c = 0; c < 8; ++c) acc[c] = z;
  float mrow[8], lrow[8];
  #pragma unroll
  for (int j = 0; j < 8; ++j) { mrow[j] = -3.0e38f; lrow[j] = 0.0f; }

  const float scale = 0.0883883476483184f;    // 1/sqrt(128)
  int ntiles = (q0 + 47) >> 5;                // causal: keys 0..q0+15

  for (int t = 0; t < ntiles; ++t) {
    int kv0 = t << 5;
    v8f s0 = z, s1 = z;
    #pragma unroll
    for (int c = 0; c < 4; ++c) {
      v16bf b0 = load_b(K + ((size_t)hk * SEQ + kv0) * HD + c * 32, HD, lane);
      v16bf b1 = load_b(K + ((size_t)hk * SEQ + kv0 + 16) * HD + c * 32, HD, lane);
      s0 = wmma_bf16(qf[c], b0, s0);
      s1 = wmma_bf16(qf[c], b1, s1);
    }
    int n0i = kv0 + l15, n1i = n0i + 16;
    int msk0 = amask[n0i], msk1 = amask[n1i];
    #pragma unroll
    for (int j = 0; j < 8; ++j) {
      int m = q0 + mb + j;
      float v0 = s0[j] * scale;
      float v1 = s1[j] * scale;
      if (n0i > m || msk0 == 0) v0 = -3.0e38f;
      if (n1i > m || msk1 == 0) v1 = -3.0e38f;
      float mx = fmaxf(v0, v1);
      #pragma unroll
      for (int o = 1; o < 16; o <<= 1)        // row lives in one 16-lane half
        mx = fmaxf(mx, __shfl_xor(mx, o, 32));
      float mnew = fmaxf(mrow[j], mx);
      float a0 = (v0 <= -1.0e38f) ? 0.0f : __expf(v0 - mnew);
      float a1 = (v1 <= -1.0e38f) ? 0.0f : __expf(v1 - mnew);
      float corr = __expf(mrow[j] - mnew);
      float rs = a0 + a1;
      #pragma unroll
      for (int o = 1; o < 16; o <<= 1)
        rs += __shfl_xor(rs, o, 32);
      lrow[j] = lrow[j] * corr + rs;
      mrow[j] = mnew;
      #pragma unroll
      for (int c = 0; c < 8; ++c) acc[c][j] *= corr;
      Plds[wib][mb + j][l15]      = f2bf(a0); // C-layout -> row-major scratch
      Plds[wib][mb + j][l15 + 16] = f2bf(a1);
    }
    asm volatile("s_wait_dscnt 0x0" ::: "memory");
    v16bf pf = load_a(&Plds[wib][0][0], 32, lane);  // row-major -> A fragment
    #pragma unroll
    for (int c = 0; c < 8; ++c) {
      v16bf vb = load_b(Vt + ((size_t)hk * HD + c * 16) * SEQ + kv0, SEQ, lane);
      acc[c] = wmma_bf16(pf, vb, acc[c]);
    }
  }

  #pragma unroll
  for (int j = 0; j < 8; ++j) {
    float inv = (lrow[j] > 0.0f) ? (1.0f / lrow[j]) : 0.0f;
    int m = q0 + mb + j;
    #pragma unroll
    for (int c = 0; c < 8; ++c)
      O[(size_t)m * HIDDEN + h * HD + c * 16 + l15] = f2bf(acc[c][j] * inv);
  }
}

// ---------------- output projection (fp32 result) ----------------
__global__ void __launch_bounds__(256) gemm_out_kernel(
    const unsigned short* __restrict__ O, const unsigned short* __restrict__ WoT,
    float* __restrict__ out) {
  __shared__ __align__(16) unsigned short Bst[2][64][32];
  int tid = threadIdx.x, lane = tid & 31, wib = tid >> 5;
  int bm = blockIdx.x / (HIDDEN / 64);                   // 16 m-blocks
  int bn = blockIdx.x % (HIDDEN / 64);                   // 64 n-groups
  int m0 = bm * 128 + wib * 16;
  int n0 = bn * 64;

  v8f acc[4];
  v8f z = {};
  #pragma unroll
  for (int t = 0; t < 4; ++t) acc[t] = z;

  gemm_wave_16x64<HIDDEN>(O + (size_t)m0 * HIDDEN, WoT + (size_t)n0 * HIDDEN,
                          &Bst[0][0][0], lane, tid, acc);

  int l15 = lane & 15;
  int mb  = (lane >> 4) << 3;
  #pragma unroll
  for (int t = 0; t < 4; ++t) {
    #pragma unroll
    for (int j = 0; j < 8; ++j)
      out[(size_t)(m0 + mb + j) * HIDDEN + n0 + t * 16 + l15] = acc[t][j];
  }
}

extern "C" void kernel_launch(void* const* d_in, const int* in_sizes, int n_in,
                              void* d_out, int out_size, void* d_ws, size_t ws_size,
                              hipStream_t stream) {
  (void)in_sizes; (void)n_in; (void)out_size; (void)ws_size;
  const float* X     = (const float*)d_in[0];
  const int*   amask = (const int*)d_in[1];
  const int*   pos   = (const int*)d_in[2];
  const float* Wq    = (const float*)d_in[3];
  const float* Wk    = (const float*)d_in[4];
  const float* Wv    = (const float*)d_in[5];
  const float* Wo    = (const float*)d_in[6];
  float* out = (float*)d_out;

  char* ws = (char*)d_ws;
  size_t o = 0;
  unsigned short* Xb  = (unsigned short*)(ws + o); o += (size_t)SEQ * HIDDEN * 2;   // 16 MiB
  unsigned short* Wt  = (unsigned short*)(ws + o); o += (size_t)NQKV * HIDDEN * 2;  // 48 MiB
  unsigned short* WoT = (unsigned short*)(ws + o); o += (size_t)HIDDEN * HIDDEN * 2;// 32 MiB
  unsigned short* Qb  = (unsigned short*)(ws + o); o += (size_t)NH * SEQ * HD * 2;  // 16 MiB
  unsigned short* Kb  = (unsigned short*)(ws + o); o += (size_t)NKV * SEQ * HD * 2; //  4 MiB
  unsigned short* Vtb = (unsigned short*)(ws + o); o += (size_t)NKV * SEQ * HD * 2; //  4 MiB
  unsigned short* Ob  = (unsigned short*)(ws + o); o += (size_t)SEQ * HIDDEN * 2;   // 16 MiB

  // 1) precision conversion + weight transposition (bf16, N-major)
  cvt_x_kernel<<<(SEQ * HIDDEN) / 256, 256, 0, stream>>>(X, Xb, SEQ * HIDDEN);
  cvt_wT_kernel<<<(HIDDEN * 4096) / 256, 256, 0, stream>>>(Wq, Wt, HIDDEN, 4096);
  cvt_wT_kernel<<<(HIDDEN * 1024) / 256, 256, 0, stream>>>(Wk, Wt + (size_t)4096 * HIDDEN, HIDDEN, 1024);
  cvt_wT_kernel<<<(HIDDEN * 1024) / 256, 256, 0, stream>>>(Wv, Wt + (size_t)5120 * HIDDEN, HIDDEN, 1024);
  cvt_wT_kernel<<<(HIDDEN * HIDDEN) / 256, 256, 0, stream>>>(Wo, WoT, HIDDEN, HIDDEN);

  // 2) fused QKV projection (async-LDS staged, WMMA bf16)
  gemm_qkv_kernel<<<(SEQ / 128) * (NQKV / 64), 256, 0, stream>>>(Xb, Wt, Qb, Kb, Vtb);

  // 3) RoPE on Q and K
  rope_kernel<<<(NH + NKV) * SEQ, 64, 0, stream>>>(Qb, Kb, pos);

  // 4) flash attention (WMMA bf16, online softmax)
  flash_attn_kernel<<<NH * (SEQ / 16) / 4, 128, 0, stream>>>(Qb, Kb, Vtb, amask, Ob);

  // 5) output projection -> fp32
  gemm_out_kernel<<<(SEQ / 128) * (HIDDEN / 64), 256, 0, stream>>>(Ob, WoT, out);
}
